// Decoder_21758304321955
// MI455X (gfx1250) — matrix operand order repaired and checked by reading, not compiled, not run on hardware
//
#include <hip/hip_runtime.h>
#include <hip/hip_bf16.h>

// Problem dims (fixed by the reference)
#define BATCH   64
#define EMB     512
#define HID     1024
#define SEQ     256
#define G4H     4096    // 4*HID
#define SRCLEN  128
#define VOCAB   32000

// ---------------------------------------------------------------------------
// WMMA fragment types (CDNA5 wave32, 16x16x32 bf16 -> f32)
// ---------------------------------------------------------------------------
typedef __attribute__((ext_vector_type(16))) __bf16 bf16x16;
typedef __attribute__((ext_vector_type(8)))  float  f32x8;
typedef __attribute__((ext_vector_type(4)))  int    i32x4;

union FragU { i32x4 q[2]; bf16x16 v; };

// Load a 16-element bf16 fragment as two 16-byte chunks (may be disjoint)
__device__ __forceinline__ bf16x16 load_frag_split(const __bf16* p0, const __bf16* p1) {
  FragU f;
  f.q[0] = *reinterpret_cast<const i32x4*>(p0);
  f.q[1] = *reinterpret_cast<const i32x4*>(p1);
  return f.v;
}

__device__ __forceinline__ float sigmoid_f(float x) {
  return 1.0f / (1.0f + __expf(-x));
}
__device__ __forceinline__ float tanh_f(float x) {
  float e = __expf(2.0f * x);
  return 1.0f - 2.0f / (e + 1.0f);   // stable at +/- inf
}

// CDNA5 async global->LDS path (ASYNCcnt-tracked, no VGPR round-trip).
// Builtin signature (from compiler diagnostics): pointers are v4i-typed,
// global source in AS(1), LDS destination in AS(3), then imm offset, imm cpol.
#if __has_builtin(__builtin_amdgcn_global_load_async_to_lds_b128) && \
    __has_builtin(__builtin_amdgcn_s_wait_asynccnt)
#define USE_ASYNC_LDS 1
#else
#define USE_ASYNC_LDS 0
#endif

#if USE_ASYNC_LDS
#define ASYNC_B128(gsrc, ldst, off)                                         \
  __builtin_amdgcn_global_load_async_to_lds_b128(                           \
      (__attribute__((address_space(1))) i32x4*)(gsrc),                     \
      (__attribute__((address_space(3))) i32x4*)(ldst), (off), 0)
#endif

// ---------------------------------------------------------------------------
// Setup kernel 1: convert & pack weights into WMMA-B-tile-major bf16 layout.
// Logical weight: Wcat[n][k], n in [0,4096), k = [W_ih | W_hh] concat.
// Packed layout: [n_tile][k_tile][lane*16 + dword*2 + h] (1KB per 32x16 tile),
// matching the assumed 16-bit B-matrix VGPR striping:
//   lanes 0-15 : N = lane,    K = kt*32 + 2*j + h + 8*(j>>2)      (j = dword 0..7)
//   lanes 16-31: N = lane-16, K = kt*32 + 8 + 2*j + h + 8*(j>>2)
// ---------------------------------------------------------------------------
__global__ void pack_weights_k(const float* __restrict__ Wih,
                               const float* __restrict__ Whh,
                               int kih, int khh, int ktiles,
                               __bf16* __restrict__ dst) {
  long idx = (long)blockIdx.x * blockDim.x + threadIdx.x;  // one dword (2 elems)
  long total = (long)256 * ktiles * 256;                   // 256 n_tiles * kt * 32 lanes * 8 dwords
  if (idx >= total) return;
  int j    = (int)(idx & 7);
  int lane = (int)((idx >> 3) & 31);
  int kt   = (int)((idx >> 8) % ktiles);
  int nt   = (int)(idx / ((long)ktiles << 8));
  int n     = nt * 16 + (lane & 15);
  int kbase = kt * 32 + ((lane >> 4) << 3) + (j << 1) + ((j >> 2) << 3);
  __bf16* out = dst + (((long)nt * ktiles + kt) << 9) + lane * 16 + (j << 1);
#pragma unroll
  for (int h = 0; h < 2; ++h) {
    int k = kbase + h;
    float v = (k < kih) ? Wih[(long)n * kih + k]
                        : Whh[(long)n * khh + (k - kih)];
    out[h] = (__bf16)v;
  }
}

// ---------------------------------------------------------------------------
// Setup kernel 2: gather token embeddings -> bf16, layout [s][b][e]
// ---------------------------------------------------------------------------
__global__ void gather_embed_k(const int* __restrict__ tokens,   // (B,S)
                               const float* __restrict__ embed,  // (V,E)
                               __bf16* __restrict__ X) {
  int sb = blockIdx.x;            // s*64 + b
  int s = sb >> 6, b = sb & 63;
  int tok = tokens[b * SEQ + s];
  const float* src = embed + (long)tok * EMB;
  __bf16* dst = X + (long)sb * EMB;
  for (int e = threadIdx.x; e < EMB; e += blockDim.x)
    dst[e] = (__bf16)src[e];
}

// ---------------------------------------------------------------------------
// Setup kernel 3: time-invariant rows [context | lang_embed] -> bf16 (B, 2E)
// ---------------------------------------------------------------------------
__global__ void build_static_k(const float* __restrict__ enc_outs,  // (B,SRC,E)
                               const int* __restrict__ langs,       // (B,)
                               const float* __restrict__ embed,
                               __bf16* __restrict__ st) {
  int b = blockIdx.x;
  const float* ctx = enc_outs + ((long)b * SRCLEN + (SRCLEN - 1)) * EMB;
  const float* lem = embed + (long)langs[b] * EMB;
  __bf16* dst = st + (long)b * (2 * EMB);
  for (int e = threadIdx.x; e < EMB; e += blockDim.x) {
    dst[e]       = (__bf16)ctx[e];
    dst[EMB + e] = (__bf16)lem[e];
  }
}

// ---------------------------------------------------------------------------
// Setup kernel 4: initial h (bf16) / c (f32) from encoder state
// ---------------------------------------------------------------------------
__global__ void init_state_k(const float* __restrict__ hid,  // (2,B,H)
                             const float* __restrict__ cel,  // (2,B,H)
                             __bf16* __restrict__ h0, __bf16* __restrict__ h1,
                             float* __restrict__ c0, float* __restrict__ c1) {
  int idx = blockIdx.x * blockDim.x + threadIdx.x;  // 0 .. 2*B*H-1
  int layer = idx >> 16;
  int off   = idx & 65535;
  if (layer == 0) { h0[off] = (__bf16)hid[idx]; c0[off] = cel[idx]; }
  else            { h1[off] = (__bf16)hid[idx]; c1[off] = cel[idx]; }
}

__global__ void combine_bias_k(const float* __restrict__ bih0, const float* __restrict__ bhh0,
                               const float* __restrict__ bih1, const float* __restrict__ bhh1,
                               float* __restrict__ bias0, float* __restrict__ bias1) {
  int i = blockIdx.x * blockDim.x + threadIdx.x;
  if (i < G4H) { bias0[i] = bih0[i] + bhh0[i]; bias1[i] = bih1[i] + bhh1[i]; }
}

// ---------------------------------------------------------------------------
// Fused per-step LSTM cell:
//   gates = [A-regions] @ Wcat^T + bias ; then elementwise cell in-register.
// Grid: 64 blocks (one per 16-wide hidden-unit slice), 128 threads (4 waves).
// Wave w handles M-tile w (batch rows 16w..16w+15) and all 4 gate tiles, so
// the sigmoid/tanh cell update closes over i,f,g,o without touching memory.
//
// Weight tiles are shared by all 4 waves of the WG, so they are staged into
// LDS with double-buffered GLOBAL_LOAD_ASYNC_TO_LDS_B128 (ASYNCcnt) when the
// toolchain exposes the builtins; otherwise each wave loads B directly.
//
// A-fragment lane mapping (16-bit A 16x32):
//   M = lane&15; lanes 0-15 hold K {0..7,16..23}, lanes 16-31 {8..15,24..31}.
// C/D mapping: vgpr r, lane l -> (M = r + 8*(l>>4), N = l&15).
// ---------------------------------------------------------------------------
__global__ __launch_bounds__(128) void lstm_step_k(
    const __bf16* __restrict__ a0, int l0,   // region 0 rows [b][l0]
    const __bf16* __restrict__ a1, int l1,   // region 1
    const __bf16* __restrict__ a2, int l2,   // region 2 (may be unused)
    const __bf16* __restrict__ Wp, int ktiles,
    const float*  __restrict__ bias,
    const float*  __restrict__ c_prev, float* __restrict__ c_new,
    __bf16* __restrict__ h_new, float* __restrict__ out_f32) {
  const int tid   = threadIdx.x;
  const int lane  = tid & 31;
  const int mtile = tid >> 5;              // 0..3
  const int u0    = blockIdx.x << 4;       // hidden-unit slice base

  const int mrow = lane & 15;
  const int koff = (lane >> 4) << 3;       // 0 or 8
  const int bA   = mtile * 16 + mrow;      // batch row for A loads
  const int s01  = l0 + l1;

  f32x8 acc[4] = {};

#if USE_ASYNC_LDS
  // double-buffered staging of the 4 gate tiles (4KB per K-chunk)
  __shared__ __align__(16) __bf16 Bsm[2][4 * 512];
  // this thread stages 32B of gate `mtile`'s tile each chunk
  const __bf16* wsrc0 = Wp + ((((long)(mtile << 6) + blockIdx.x) * ktiles) << 9) + lane * 16;
  __bf16* ldst0 = &Bsm[0][mtile * 512 + lane * 16];
  __bf16* ldst1 = &Bsm[1][mtile * 512 + lane * 16];
  // prologue: stage kt = 0 into buffer 0
  ASYNC_B128(wsrc0, ldst0, 0);
  ASYNC_B128(wsrc0 + 8, ldst0 + 8, 0);
#endif

  for (int kt = 0; kt < ktiles; ++kt) {
    const int ks = kt << 5;
    // region select is uniform per 32-wide chunk (boundaries are multiples of 32)
    const __bf16* base; int rs, rl;
    if (ks < l0)       { base = a0; rs = 0;   rl = l0; }
    else if (ks < s01) { base = a1; rs = l0;  rl = l1; }
    else               { base = a2; rs = s01; rl = l2; }
    const __bf16* arow = base + (long)bA * rl + (ks - rs) + koff;

#if USE_ASYNC_LDS
    const int cur = kt & 1;
    __builtin_amdgcn_s_wait_asynccnt(0);   // my stores for buf[cur] landed
    __syncthreads();                       // everyone's stores landed; prior reads done
    if (kt + 1 < ktiles) {                 // stage next chunk into the other buffer
      const __bf16* src = wsrc0 + ((long)(kt + 1) << 9);
      __bf16* dst = (cur == 0) ? ldst1 : ldst0;
      ASYNC_B128(src, dst, 0);
      ASYNC_B128(src + 8, dst + 8, 0);
    }
    bf16x16 afrag = load_frag_split(arow, arow + 16);
#pragma unroll
    for (int g = 0; g < 4; ++g) {
      const __bf16* bp = &Bsm[cur][g * 512 + lane * 16];
      bf16x16 bfrag = load_frag_split(bp, bp + 8);      // ds_load_b128 x2
      acc[g] = __builtin_amdgcn_wmma_f32_16x16x32_bf16(
          false, afrag, false, bfrag, (short)0, acc[g], false, false);
    }
#else
    bf16x16 afrag = load_frag_split(arow, arow + 16);
    const __bf16* wbase = Wp + ((long)kt << 9) + lane * 16;
#pragma unroll
    for (int g = 0; g < 4; ++g) {
      const int nt = (g << 6) + blockIdx.x;             // gate-major N tiles
      const __bf16* wp = wbase + (((long)nt * ktiles) << 9);
      bf16x16 bfrag = load_frag_split(wp, wp + 8);
      if (kt + 2 < ktiles)
        __builtin_prefetch(wp + (2 << 9), 0, 3);        // next+2 K-tile of this gate
      acc[g] = __builtin_amdgcn_wmma_f32_16x16x32_bf16(
          false, afrag, false, bfrag, (short)0, acc[g], false, false);
    }
#endif
  }

  // ---- fused LSTM cell (in-register gates) ----
  const int u   = u0 + (lane & 15);
  const float bi = bias[u];
  const float bf = bias[HID + u];
  const float bg = bias[2 * HID + u];
  const float bo = bias[3 * HID + u];
  const int rbase = mtile * 16 + ((lane >> 4) << 3);
#pragma unroll
  for (int r = 0; r < 8; ++r) {
    const int  b   = rbase + r;
    const long idx = (long)b * HID + u;
    float gi = sigmoid_f(acc[0][r] + bi);
    float gf = sigmoid_f(acc[1][r] + bf);
    float gg = tanh_f  (acc[2][r] + bg);
    float go = sigmoid_f(acc[3][r] + bo);
    float cn = gf * c_prev[idx] + gi * gg;
    c_new[idx] = cn;
    float h = go * tanh_f(cn);
    h_new[idx] = (__bf16)h;
    if (out_f32) out_f32[idx] = h;
  }
}

// ---------------------------------------------------------------------------
// Host driver
// ---------------------------------------------------------------------------
extern "C" void kernel_launch(void* const* d_in, const int* in_sizes, int n_in,
                              void* d_out, int out_size, void* d_ws, size_t ws_size,
                              hipStream_t stream) {
  const int*   tokens  = (const int*)d_in[0];
  const int*   langs   = (const int*)d_in[1];
  const float* encOuts = (const float*)d_in[2];
  const float* encHid  = (const float*)d_in[3];
  const float* encCel  = (const float*)d_in[4];
  const float* embed   = (const float*)d_in[5];
  const float* Wih0 = (const float*)d_in[6];
  const float* Whh0 = (const float*)d_in[7];
  const float* bih0 = (const float*)d_in[8];
  const float* bhh0 = (const float*)d_in[9];
  const float* Wih1 = (const float*)d_in[10];
  const float* Whh1 = (const float*)d_in[11];
  const float* bih1 = (const float*)d_in[12];
  const float* bhh1 = (const float*)d_in[13];
  float* out = (float*)d_out;

  // workspace carve-up (all offsets 256B aligned)
  char* ws = (char*)d_ws;
  size_t off = 0;
  auto alloc = [&](size_t bytes) -> void* {
    void* p = ws + off;
    off += (bytes + 255) & ~(size_t)255;
    return p;
  };
  __bf16* Xemb = (__bf16*)alloc((size_t)SEQ * BATCH * EMB * 2);   // 16 MB
  __bf16* Wp0  = (__bf16*)alloc((size_t)G4H * 2560 * 2);          // 21 MB
  __bf16* Wp1  = (__bf16*)alloc((size_t)G4H * 2048 * 2);          // 17 MB
  __bf16* Stat = (__bf16*)alloc((size_t)BATCH * 2 * EMB * 2);
  float*  bias0 = (float*)alloc((size_t)G4H * 4);
  float*  bias1 = (float*)alloc((size_t)G4H * 4);
  __bf16* h0[2]; __bf16* h1[2]; float* c0[2]; float* c1[2];
  for (int i = 0; i < 2; ++i) {
    h0[i] = (__bf16*)alloc((size_t)BATCH * HID * 2);
    h1[i] = (__bf16*)alloc((size_t)BATCH * HID * 2);
    c0[i] = (float*)alloc((size_t)BATCH * HID * 4);
    c1[i] = (float*)alloc((size_t)BATCH * HID * 4);
  }

  const int KT0 = 2560 / 32;   // layer0 K = E + 2E + H
  const int KT1 = 2048 / 32;   // layer1 K = H + H

  { long tot = (long)256 * KT0 * 256;
    pack_weights_k<<<(int)((tot + 255) / 256), 256, 0, stream>>>(Wih0, Whh0, 3 * EMB, HID, KT0, Wp0); }
  { long tot = (long)256 * KT1 * 256;
    pack_weights_k<<<(int)((tot + 255) / 256), 256, 0, stream>>>(Wih1, Whh1, HID, HID, KT1, Wp1); }
  gather_embed_k<<<SEQ * BATCH, 256, 0, stream>>>(tokens, embed, Xemb);
  build_static_k<<<BATCH, 256, 0, stream>>>(encOuts, langs, embed, Stat);
  init_state_k<<<(2 * BATCH * HID) / 256, 256, 0, stream>>>(encHid, encCel, h0[0], h1[0], c0[0], c1[0]);
  combine_bias_k<<<(G4H + 255) / 256, 256, 0, stream>>>(bih0, bhh0, bih1, bhh1, bias0, bias1);

  // recurrence: stream ordering carries the step-to-step dependency
  for (int s = 0; s < SEQ; ++s) {
    const int rp = s & 1, wp = rp ^ 1;
    // layer 0: A = [x_t | static | h0_prev], K = 2560
    lstm_step_k<<<HID / 16, 128, 0, stream>>>(
        Xemb + (long)s * BATCH * EMB, EMB,
        Stat, 2 * EMB,
        h0[rp], HID,
        Wp0, KT0, bias0, c0[rp], c0[wp], h0[wp], (float*)nullptr);
    // layer 1: A = [h0_new | h1_prev], K = 2048 ; writes outs[s]
    lstm_step_k<<<HID / 16, 128, 0, stream>>>(
        h0[wp], HID,
        h1[rp], HID,
        (const __bf16*)nullptr, 0,
        Wp1, KT1, bias1, c1[rp], c1[wp], h1[wp],
        out + (long)s * BATCH * HID);
  }
}
